// EncoderLayer_67156108640885
// MI455X (gfx1250) — compile-verified, matrix operand order checked by reading, not run on hardware
//
#include <hip/hip_runtime.h>
#include <hip/hip_bf16.h>

// ---------------------------------------------------------------------------
// Encoder layer for MI455X (gfx1250): bf16 WMMA GEMMs + fused attention.
// D=1024, H=16, HD=64, B=4, S=1024.
// GEMM uses double-buffered async global->LDS staging (ASYNCcnt) when the
// toolchain exposes the gfx1250 async builtins.
// ---------------------------------------------------------------------------

typedef __attribute__((ext_vector_type(16))) __bf16 v16bf;
typedef __attribute__((ext_vector_type(8)))  float  v8f;
typedef int v4i __attribute__((vector_size(16)));   // matches builtin param

#define D_   1024
#define H_   16
#define HD_  64
#define B_   4
#define S_   1024
#define M_   (B_ * S_)     // 4096 rows of activations
#define EPS_ 1e-5f

#if __has_builtin(__builtin_amdgcn_global_load_async_to_lds_b128)
#define HAVE_ASYNC 1
#else
#define HAVE_ASYNC 0
#endif

typedef __attribute__((address_space(1))) v4i as1_v4i;  // global int4
typedef __attribute__((address_space(3))) v4i as3_v4i;  // LDS int4

// ---- bf16 helpers (round-to-nearest-even) ---------------------------------
__device__ __forceinline__ unsigned short f2bf_u(float f) {
    unsigned int u = __builtin_bit_cast(unsigned int, f);
    u = u + 0x7FFFu + ((u >> 16) & 1u);
    return (unsigned short)(u >> 16);
}
__device__ __forceinline__ float bf2f(unsigned short s) {
    unsigned int u = ((unsigned int)s) << 16;
    return __builtin_bit_cast(float, u);
}
// K-index mapping for 16-bit A/B WMMA fragments (ISA 7.12.2):
// half h, lane-half lh -> k = (h&8 ? 16:0) + lh*8 + (h&7)
__device__ __forceinline__ int kmap(int hh, int lhalf) {
    return ((hh & 8) ? 16 : 0) + (lhalf << 3) + (hh & 7);
}

// ---- 16B global->LDS staging ----------------------------------------------
// Async path: GLOBAL_LOAD_ASYNC_TO_LDS_B128 (ASYNCcnt).  AS3 pointer is the
// low 32 bits of the generic LDS address; AS1 is the flat address unchanged.
__device__ __forceinline__ void stage_b128(const unsigned short* g,
                                           unsigned short* l) {
#if HAVE_ASYNC
    __builtin_amdgcn_global_load_async_to_lds_b128(
        (as1_v4i*)(unsigned long long)g,
        (as3_v4i*)(unsigned int)(unsigned long long)l,
        0, 0);
#else
    *(int4*)l = *(const int4*)g;
#endif
}
__device__ __forceinline__ void wait_async_le3() {
#if HAVE_ASYNC
#if __has_builtin(__builtin_amdgcn_s_wait_asynccnt)
    __builtin_amdgcn_s_wait_asynccnt(3);
#else
    asm volatile("s_wait_asynccnt 0x3" ::: "memory");
#endif
#endif
}
__device__ __forceinline__ void wait_async_le0() {
#if HAVE_ASYNC
#if __has_builtin(__builtin_amdgcn_s_wait_asynccnt)
    __builtin_amdgcn_s_wait_asynccnt(0);
#else
    asm volatile("s_wait_asynccnt 0x0" ::: "memory");
#endif
#endif
}

// ---------------------------------------------------------------------------
// f32 -> bf16 conversion
// ---------------------------------------------------------------------------
__global__ void cvt_bf16_kernel(const float* __restrict__ in,
                                unsigned short* __restrict__ out, int n) {
    for (int i = blockIdx.x * blockDim.x + threadIdx.x; i < n;
         i += gridDim.x * blockDim.x)
        out[i] = f2bf_u(in[i]);
}

// ---------------------------------------------------------------------------
// Tiled bf16 GEMM:  C = A[M,K] @ W[K,N] + bias, with epilogue MODE:
//   MODE 0: store bf16 in [B,H,S,HD] layout (QKV projections)
//   MODE 1: store bf16 relu(val) row-major [M,N]           (FFN1)
//   MODE 2: store f32  val + resid[M,N] row-major          (out-proj, FFN2)
// Block: 256 threads (8 waves). Tile: BM=128, BN=64, BK=32.
// Each wave computes a 32x32 region = 2x2 WMMA 16x16 tiles.
// Double-buffered LDS; next K-panel streamed by async-to-LDS while WMMAs
// consume the current one.
// ---------------------------------------------------------------------------
template <int MODE>
__global__ __launch_bounds__(256) void gemm_bf16_kernel(
    const unsigned short* __restrict__ A, const unsigned short* __restrict__ W,
    const float* __restrict__ bias, const float* __restrict__ resid,
    void* __restrict__ Cout, int M, int N, int K) {
    constexpr int BM = 128, BN = 64, BK = 32;
    constexpr int ASTR = BK + 8;  // 40 halves = 80 B  (multiple of 16 B)
    constexpr int BSTR = BN + 8;  // 72 halves = 144 B (multiple of 16 B)
    __shared__ unsigned short As[2][BM * ASTR];
    __shared__ unsigned short Bs[2][BK * BSTR];

    const int tid   = threadIdx.x;
    const int wave  = tid >> 5;
    const int lane  = tid & 31;
    const int lhalf = lane >> 4;
    const int lmod  = lane & 15;
    const int wm    = (wave & 3) * 32;   // wave row offset inside block tile
    const int wn    = (wave >> 2) * 32;  // wave col offset inside block tile
    const int m0    = blockIdx.y * BM;
    const int n0    = blockIdx.x * BN;

    // Per-stage staging work (3 x b128 per thread):
    //   A panel: 128 rows x 4 chunks of 16 B -> 512 chunks, 2 per thread
    //   W panel:  32 rows x 8 chunks of 16 B -> 256 chunks, 1 per thread
    const int ar0 = tid >> 2, ac0 = tid & 3;            // chunk 0 of A
    const int ar1 = (tid + 256) >> 2, ac1 = tid & 3;    // chunk 1 of A
    const int br  = tid >> 3, bc = tid & 7;             // chunk of W

    auto stage = [&](int k0, int buf) {
        stage_b128(A + (size_t)(m0 + ar0) * K + k0 + ac0 * 8,
                   &As[buf][ar0 * ASTR + ac0 * 8]);
        stage_b128(A + (size_t)(m0 + ar1) * K + k0 + ac1 * 8,
                   &As[buf][ar1 * ASTR + ac1 * 8]);
        stage_b128(W + (size_t)(k0 + br) * N + n0 + bc * 8,
                   &Bs[buf][br * BSTR + bc * 8]);
    };

    v8f acc[2][2] = {};
    const int NK = K / BK;

    stage(0, 0);
    for (int it = 0; it < NK; ++it) {
        const int buf = it & 1;
        if (it + 1 < NK) {
            stage((it + 1) * BK, buf ^ 1);
            wait_async_le3();   // previous stage's 3 ops have landed
        } else {
            wait_async_le0();
        }
        __syncthreads();

        v16bf af[2], bf[2];
#pragma unroll
        for (int i = 0; i < 2; ++i) {
            int mrow = wm + i * 16 + lmod;
#pragma unroll
            for (int hh = 0; hh < 16; ++hh)
                af[i][hh] = __builtin_bit_cast(
                    __bf16, As[buf][mrow * ASTR + kmap(hh, lhalf)]);
        }
#pragma unroll
        for (int i = 0; i < 2; ++i) {
            int ncol = wn + i * 16 + lmod;
#pragma unroll
            for (int hh = 0; hh < 16; ++hh)
                bf[i][hh] = __builtin_bit_cast(
                    __bf16, Bs[buf][kmap(hh, lhalf) * BSTR + ncol]);
        }
#pragma unroll
        for (int i = 0; i < 2; ++i)
#pragma unroll
            for (int j = 0; j < 2; ++j)
                acc[i][j] = __builtin_amdgcn_wmma_f32_16x16x32_bf16(
                    false, af[i], false, bf[j], (short)0, acc[i][j], false,
                    false);
        __syncthreads();
    }

    // Epilogue.  C/D layout: VGPR v -> row v + 8*lhalf, col = lmod.
#pragma unroll
    for (int i = 0; i < 2; ++i) {
#pragma unroll
        for (int j = 0; j < 2; ++j) {
            int gn = n0 + wn + j * 16 + lmod;
            float bv = bias ? bias[gn] : 0.0f;
#pragma unroll
            for (int v = 0; v < 8; ++v) {
                int gm = m0 + wm + i * 16 + v + lhalf * 8;
                float val = acc[i][j][v] + bv;
                if (MODE == 0) {
                    int b = gm >> 10, s = gm & (S_ - 1);
                    int h = gn >> 6,  hd = gn & (HD_ - 1);
                    ((unsigned short*)Cout)[((((size_t)b * H_ + h) * S_ + s)
                                             << 6) + hd] = f2bf_u(val);
                } else if (MODE == 1) {
                    val = val > 0.0f ? val : 0.0f;
                    ((unsigned short*)Cout)[(size_t)gm * N + gn] = f2bf_u(val);
                } else {
                    val += resid[(size_t)gm * N + gn];
                    ((float*)Cout)[(size_t)gm * N + gn] = val;
                }
            }
        }
    }
}

// ---------------------------------------------------------------------------
// Attention: one workgroup (128 threads / 4 waves) per (b, h, 16-query tile).
// scores = mask(QK^T/8) -> softmax over 1024 keys -> ctx = P @ V.
// ---------------------------------------------------------------------------
__global__ __launch_bounds__(128) void attn_kernel(
    const unsigned short* __restrict__ Qb,  // [B,H,S,HD] bf16
    const unsigned short* __restrict__ Kb,  // [B,H,S,HD] bf16
    const unsigned short* __restrict__ Vb,  // [B,H,S,HD] bf16
    const unsigned char* __restrict__ mask, // [B,H,S,S] bool
    unsigned short* __restrict__ ctx)       // [B,S,D] bf16
{
    constexpr int QSTR = HD_ + 8;  // 72
    __shared__ unsigned short sP[16 * S_];          // scores -> attn (bf16)
    __shared__ unsigned short sQ[16 * QSTR];        // Q tile
    __shared__ unsigned short sKV[4 * 16 * QSTR];   // per-wave K tile
    __shared__ unsigned short sV[32 * QSTR];        // V chunk (32 keys)
    __shared__ float red[16 * 8];
    __shared__ float rmax[16], rsum[16];

    const int tid   = threadIdx.x;
    const int wave  = tid >> 5;
    const int lane  = tid & 31;
    const int lhalf = lane >> 4;
    const int lmod  = lane & 15;

    const int bhq   = blockIdx.x;                 // [0, B*H*(S/16))
    const int qt    = bhq & 63;
    const int h     = (bhq >> 6) & (H_ - 1);
    const int b     = bhq >> 10;
    const int qbase = qt * 16;
    const size_t headBase = ((size_t)b * H_ + h) * S_;  // row base into [.,S,HD]

    // --- stage Q tile (16 x 64 bf16) -----------------------------------
#pragma unroll
    for (int j = 0; j < 4; ++j) {
        int lin = tid + j * 128;  // 0..511 dwords
        int r = lin >> 5, c = lin & 31;
        unsigned int v = *(const unsigned int*)(
            Qb + ((headBase + qbase + r) << 6) + c * 2);
        *(unsigned int*)(&sQ[r * QSTR + c * 2]) = v;
    }
    __syncthreads();

    // A fragments of Q (k 0..31 and 32..63), reused for every key tile.
    v16bf aQ[2];
#pragma unroll
    for (int f = 0; f < 2; ++f)
#pragma unroll
        for (int hh = 0; hh < 16; ++hh)
            aQ[f][hh] = __builtin_bit_cast(
                __bf16, sQ[lmod * QSTR + f * 32 + kmap(hh, lhalf)]);

    // --- scores: each wave handles 16 key-tiles of 16 keys -------------
    unsigned short* sKVw = &sKV[wave * 16 * QSTR];
    for (int t = 0; t < 16; ++t) {
        const int key0 = (wave * 16 + t) * 16;
        // stage K tile (16 keys x 64) with this wave's 32 lanes
#pragma unroll
        for (int j = 0; j < 16; ++j) {
            int lin = lane + j * 32;  // 0..511 dwords
            int r = lin >> 5, c = lin & 31;
            unsigned int v = *(const unsigned int*)(
                Kb + ((headBase + key0 + r) << 6) + c * 2);
            *(unsigned int*)(&sKVw[r * QSTR + c * 2]) = v;
        }
        __syncthreads();

        v16bf bK[2];
#pragma unroll
        for (int f = 0; f < 2; ++f)
#pragma unroll
            for (int hh = 0; hh < 16; ++hh)
                bK[f][hh] = __builtin_bit_cast(
                    __bf16, sKVw[lmod * QSTR + f * 32 + kmap(hh, lhalf)]);

        v8f sc = {};
        sc = __builtin_amdgcn_wmma_f32_16x16x32_bf16(false, aQ[0], false,
                                                     bK[0], (short)0, sc,
                                                     false, false);
        sc = __builtin_amdgcn_wmma_f32_16x16x32_bf16(false, aQ[1], false,
                                                     bK[1], (short)0, sc,
                                                     false, false);
        // mask + scale, write to sP
        const int key = key0 + lmod;
#pragma unroll
        for (int v = 0; v < 8; ++v) {
            int m = v + lhalf * 8;
            unsigned char mb =
                mask[(((size_t)b * H_ + h) * S_ + (qbase + m)) * S_ + key];
            float val = mb ? -INFINITY : sc[v] * 0.125f;  // 1/sqrt(64)
            sP[m * S_ + key] = f2bf_u(val);
        }
        __syncthreads();
    }

    // --- softmax over 1024 keys per row (16 rows x 8 threads) ----------
    {
        const int r = tid >> 3, sub = tid & 7;
        float mx = -INFINITY;
        for (int c = sub; c < S_; c += 8)
            mx = fmaxf(mx, bf2f(sP[r * S_ + c]));
        red[r * 8 + sub] = mx;
        __syncthreads();
        if (sub == 0) {
            float m2 = red[r * 8];
            for (int i = 1; i < 8; ++i) m2 = fmaxf(m2, red[r * 8 + i]);
            rmax[r] = m2;
        }
        __syncthreads();
        const float mrow = rmax[r];
        float sm = 0.0f;
        for (int c = sub; c < S_; c += 8)
            sm += __expf(bf2f(sP[r * S_ + c]) - mrow);
        red[r * 8 + sub] = sm;
        __syncthreads();
        if (sub == 0) {
            float s2 = 0.0f;
            for (int i = 0; i < 8; ++i) s2 += red[r * 8 + i];
            rsum[r] = s2;
        }
        __syncthreads();
        const float inv = 1.0f / rsum[r];
        for (int c = sub; c < S_; c += 8)
            sP[r * S_ + c] =
                f2bf_u(__expf(bf2f(sP[r * S_ + c]) - mrow) * inv);
        __syncthreads();
    }

    // --- ctx = P[16,1024] @ V[1024,64]; wave w owns hd cols [w*16, w*16+16)
    v8f cacc = {};
    for (int kc = 0; kc < 32; ++kc) {
        const int kbase = kc * 32;
        // stage V chunk (32 keys x 64) cooperatively: 1024 dwords / 128 thr
#pragma unroll
        for (int j = 0; j < 8; ++j) {
            int lin = tid + j * 128;
            int r = lin >> 5, c = lin & 31;
            unsigned int v = *(const unsigned int*)(
                Vb + ((headBase + kbase + r) << 6) + c * 2);
            *(unsigned int*)(&sV[r * QSTR + c * 2]) = v;
        }
        __syncthreads();

        v16bf aP, bV;
#pragma unroll
        for (int hh = 0; hh < 16; ++hh) {
            int kk = kmap(hh, lhalf);
            aP[hh] = __builtin_bit_cast(__bf16, sP[lmod * S_ + kbase + kk]);
            bV[hh] = __builtin_bit_cast(__bf16, sV[kk * QSTR + wave * 16 + lmod]);
        }
        cacc = __builtin_amdgcn_wmma_f32_16x16x32_bf16(false, aP, false, bV,
                                                       (short)0, cacc, false,
                                                       false);
        __syncthreads();
    }
    // write ctx in [B,S,D] layout
#pragma unroll
    for (int v = 0; v < 8; ++v) {
        int m = v + lhalf * 8;
        int s = qbase + m;
        int dcol = h * HD_ + wave * 16 + lmod;
        ctx[((size_t)b * S_ + s) * D_ + dcol] = f2bf_u(cacc[v]);
    }
}

// ---------------------------------------------------------------------------
// LayerNorm over last dim (1024). One block (256 thr) per row.
// Optionally writes a bf16 shadow copy (input for next GEMM).
// ---------------------------------------------------------------------------
template <bool WRITE_BF>
__global__ __launch_bounds__(256) void layernorm_kernel(
    const float* __restrict__ in, const float* __restrict__ g,
    const float* __restrict__ beta, float* __restrict__ outf,
    unsigned short* __restrict__ outb) {
    __shared__ float red[256];
    const int tid = threadIdx.x;
    const float* x = in + (size_t)blockIdx.x * D_;

    float s = 0.0f;
    for (int c = tid; c < D_; c += 256) s += x[c];
    red[tid] = s;
    __syncthreads();
    for (int st = 128; st > 0; st >>= 1) {
        if (tid < st) red[tid] += red[tid + st];
        __syncthreads();
    }
    const float mu = red[0] * (1.0f / D_);
    __syncthreads();

    float vs = 0.0f;
    for (int c = tid; c < D_; c += 256) {
        float d = x[c] - mu;
        vs += d * d;
    }
    red[tid] = vs;
    __syncthreads();
    for (int st = 128; st > 0; st >>= 1) {
        if (tid < st) red[tid] += red[tid + st];
        __syncthreads();
    }
    const float inv = rsqrtf(red[0] * (1.0f / D_) + EPS_);

    for (int c = tid; c < D_; c += 256) {
        float o = (x[c] - mu) * inv * g[c] + beta[c];
        outf[(size_t)blockIdx.x * D_ + c] = o;
        if (WRITE_BF) outb[(size_t)blockIdx.x * D_ + c] = f2bf_u(o);
    }
}

// ---------------------------------------------------------------------------
// Host-side orchestration.
// ---------------------------------------------------------------------------
static inline size_t ws_alloc(size_t& off, size_t bytes) {
    size_t o = off;
    off = (off + bytes + 255) & ~(size_t)255;
    return o;
}

extern "C" void kernel_launch(void* const* d_in, const int* in_sizes, int n_in,
                              void* d_out, int out_size, void* d_ws,
                              size_t ws_size, hipStream_t stream) {
    (void)in_sizes; (void)n_in; (void)out_size; (void)ws_size;
    const float* x    = (const float*)d_in[0];
    const float* y    = (const float*)d_in[1];
    const unsigned char* mask = (const unsigned char*)d_in[2];  // jnp bool
    const float* Wq = (const float*)d_in[3];
    const float* bq = (const float*)d_in[4];
    const float* Wk = (const float*)d_in[5];
    const float* bk = (const float*)d_in[6];
    const float* Wv = (const float*)d_in[7];
    const float* bv = (const float*)d_in[8];
    const float* Wo = (const float*)d_in[9];
    const float* bo = (const float*)d_in[10];
    const float* W1 = (const float*)d_in[11];
    const float* b1 = (const float*)d_in[12];
    const float* W2 = (const float*)d_in[13];
    const float* b2 = (const float*)d_in[14];
    const float* g1    = (const float*)d_in[15];
    const float* beta1 = (const float*)d_in[16];
    const float* g2    = (const float*)d_in[17];
    const float* beta2 = (const float*)d_in[18];

    char* ws = (char*)d_ws;
    size_t off = 0;
    unsigned short* xb  = (unsigned short*)(ws + ws_alloc(off, (size_t)M_ * D_ * 2));
    unsigned short* yb  = (unsigned short*)(ws + ws_alloc(off, (size_t)M_ * D_ * 2));
    unsigned short* Wqb = (unsigned short*)(ws + ws_alloc(off, (size_t)D_ * D_ * 2));
    unsigned short* Wkb = (unsigned short*)(ws + ws_alloc(off, (size_t)D_ * D_ * 2));
    unsigned short* Wvb = (unsigned short*)(ws + ws_alloc(off, (size_t)D_ * D_ * 2));
    unsigned short* Wob = (unsigned short*)(ws + ws_alloc(off, (size_t)D_ * D_ * 2));
    unsigned short* W1b = (unsigned short*)(ws + ws_alloc(off, (size_t)D_ * 4 * D_ * 2));
    unsigned short* W2b = (unsigned short*)(ws + ws_alloc(off, (size_t)4 * D_ * D_ * 2));
    unsigned short* Qb  = (unsigned short*)(ws + ws_alloc(off, (size_t)M_ * D_ * 2));
    unsigned short* Kb  = (unsigned short*)(ws + ws_alloc(off, (size_t)M_ * D_ * 2));
    unsigned short* Vb  = (unsigned short*)(ws + ws_alloc(off, (size_t)M_ * D_ * 2));
    unsigned short* ctxb= (unsigned short*)(ws + ws_alloc(off, (size_t)M_ * D_ * 2));
    float*          t1  = (float*)(ws + ws_alloc(off, (size_t)M_ * D_ * 4));
    float*          x1f = (float*)(ws + ws_alloc(off, (size_t)M_ * D_ * 4));
    unsigned short* x1b = (unsigned short*)(ws + ws_alloc(off, (size_t)M_ * D_ * 2));
    unsigned short* h1  = (unsigned short*)(ws + ws_alloc(off, (size_t)M_ * 4 * D_ * 2));
    float*          t2  = t1;  // t1 is consumed by LN1 before FFN2 writes t2

    // --- bf16 conversions ------------------------------------------------
    auto cvt = [&](const float* src, unsigned short* dst, int n) {
        int blocks = (n + 255) / 256;
        if (blocks > 8192) blocks = 8192;
        cvt_bf16_kernel<<<blocks, 256, 0, stream>>>(src, dst, n);
    };
    cvt(x,  xb,  M_ * D_);
    cvt(y,  yb,  M_ * D_);
    cvt(Wq, Wqb, D_ * D_);
    cvt(Wk, Wkb, D_ * D_);
    cvt(Wv, Wvb, D_ * D_);
    cvt(Wo, Wob, D_ * D_);
    cvt(W1, W1b, D_ * 4 * D_);
    cvt(W2, W2b, 4 * D_ * D_);

    // --- QKV projections (MODE 0 -> [B,H,S,HD] bf16) ---------------------
    dim3 gP(D_ / 64, M_ / 128);
    gemm_bf16_kernel<0><<<gP, 256, 0, stream>>>(xb, Wqb, bq, nullptr, Qb, M_, D_, D_);
    gemm_bf16_kernel<0><<<gP, 256, 0, stream>>>(yb, Wkb, bk, nullptr, Kb, M_, D_, D_);
    gemm_bf16_kernel<0><<<gP, 256, 0, stream>>>(yb, Wvb, bv, nullptr, Vb, M_, D_, D_);

    // --- attention -------------------------------------------------------
    attn_kernel<<<B_ * H_ * (S_ / 16), 128, 0, stream>>>(Qb, Kb, Vb, mask, ctxb);

    // --- output projection + residual (f32) ------------------------------
    gemm_bf16_kernel<2><<<gP, 256, 0, stream>>>(ctxb, Wob, bo, x, t1, M_, D_, D_);
    layernorm_kernel<true><<<M_, 256, 0, stream>>>(t1, g1, beta1, x1f, x1b);

    // --- FFN -------------------------------------------------------------
    dim3 gF1(4 * D_ / 64, M_ / 128);
    gemm_bf16_kernel<1><<<gF1, 256, 0, stream>>>(x1b, W1b, b1, nullptr, h1, M_, 4 * D_, D_);
    gemm_bf16_kernel<2><<<gP, 256, 0, stream>>>(h1, W2b, b2, x1f, t2, M_, D_, 4 * D_);
    layernorm_kernel<false><<<M_, 256, 0, stream>>>(t2, g2, beta2, (float*)d_out, nullptr);
}
